// TransformerModel_64218351009958
// MI455X (gfx1250) — matrix operand order repaired and checked by reading, not compiled, not run on hardware
//
#include <hip/hip_runtime.h>
#include <hip/hip_bf16.h>

// ---------------- model constants ----------------
#define VV    32000
#define DD    1024
#define HH    16
#define HD    64
#define NLYR  4
#define WIN   256
#define LL    2048
#define II    3280
#define IPAD  3296      // II rounded up to multiple of 32 (K-dim for w2 GEMM)
#define EPSF  1e-5f

// GEMM register blocking: each wave computes (MB*16) x (NB*16) of C.
#define MB 2            // M must be a multiple of 32 (LL = 2048: ok)
#define NB 4

typedef __bf16 bf16;
typedef __attribute__((ext_vector_type(16))) __bf16 bf16x16;
typedef __attribute__((ext_vector_type(8)))  __bf16 bf16x8;
typedef __attribute__((ext_vector_type(8)))  float  f32x8;

// =====================================================================
// fp32 -> bf16 convert with optional K-padding (rows x Kp, zero tail)
// =====================================================================
__global__ __launch_bounds__(256)
void cvt_bf16_pad(const float* __restrict__ src, bf16* __restrict__ dst,
                  int K, int Kp, long long total) {
    long long i = (long long)blockIdx.x * 256 + threadIdx.x;
    if (i >= total) return;
    long long r = i / Kp;
    int kp = (int)(i - r * Kp);
    float v = (kp < K) ? src[r * (long long)K + kp] : 0.0f;
    dst[i] = (bf16)v;
}

// =====================================================================
// Embedding gather: x[l,d] = embed_w[tokens[l]*D + d]  (fp32)
// =====================================================================
__global__ __launch_bounds__(256)
void embed_k(const int* __restrict__ tok, const float* __restrict__ emb,
             float* __restrict__ x, long long total) {
    long long i = (long long)blockIdx.x * 256 + threadIdx.x;
    if (i >= total) return;
    long long l = i / DD;
    int d = (int)(i - l * DD);
    x[i] = emb[(long long)tok[l] * DD + d];
}

// =====================================================================
// RMSNorm over D=1024 per row; clamp(+-1e4), eps guards; writes bf16
// One 256-thread block per row, 4 elements/thread.
// =====================================================================
__global__ __launch_bounds__(256)
void rmsnorm_row(const float* __restrict__ x, const float* __restrict__ w,
                 bf16* __restrict__ out) {
    __shared__ float red[8];
    int row = blockIdx.x;
    const float* xr = x + (size_t)row * DD;
    float vals[4];
    float ss = 0.0f;
#pragma unroll
    for (int i = 0; i < 4; ++i) {
        float t = xr[threadIdx.x + 256 * i];
        t = fminf(fmaxf(t, -10000.0f), 10000.0f);
        vals[i] = t;
        ss += t * t;
    }
#pragma unroll
    for (int off = 16; off >= 1; off >>= 1) ss += __shfl_xor(ss, off, 32);
    if ((threadIdx.x & 31) == 0) red[threadIdx.x >> 5] = ss;
    __syncthreads();
    float tot = 0.0f;
#pragma unroll
    for (int i = 0; i < 8; ++i) tot += red[i];
    float inv = 1.0f / sqrtf(fmaxf(tot * (1.0f / (float)DD), EPSF) + EPSF);
#pragma unroll
    for (int i = 0; i < 4; ++i) {
        int c = threadIdx.x + 256 * i;
        float o = vals[i] * inv * w[c];
        if (!__builtin_isfinite(o)) o = 0.0f;
        out[(size_t)row * DD + c] = (bf16)o;
    }
}

// =====================================================================
// Per-head RMSNorm over HD=64 vectors, in place (q / k). One wave/row.
// =====================================================================
__global__ __launch_bounds__(256)
void head_rmsnorm(float* __restrict__ qk, const float* __restrict__ w, int nrows) {
    int wave = blockIdx.x * 8 + (threadIdx.x >> 5);
    if (wave >= nrows) return;
    int lane = threadIdx.x & 31;
    float* r = qk + (size_t)wave * HD;
    float a = fminf(fmaxf(r[lane],      -10000.0f), 10000.0f);
    float b = fminf(fmaxf(r[lane + 32], -10000.0f), 10000.0f);
    float ss = a * a + b * b;
#pragma unroll
    for (int off = 16; off >= 1; off >>= 1) ss += __shfl_xor(ss, off, 32);
    float inv = 1.0f / sqrtf(fmaxf(ss * (1.0f / (float)HD), EPSF) + EPSF);
    float oa = a * inv * w[lane];
    float ob = b * inv * w[lane + 32];
    if (!__builtin_isfinite(oa)) oa = 0.0f;
    if (!__builtin_isfinite(ob)) ob = 0.0f;
    r[lane] = oa;
    r[lane + 32] = ob;
}

// =====================================================================
// Sliding-window causal attention, online softmax.
// One wave per (query l, head h); lanes own 2 dims of HD=64.
// Writes bf16 output directly (feeds the wo GEMM).
// =====================================================================
__global__ __launch_bounds__(256)
void attn_sw(const float* __restrict__ q, const float* __restrict__ k,
             const float* __restrict__ v, bf16* __restrict__ out_bf) {
    int wave = blockIdx.x * 8 + (threadIdx.x >> 5);
    int l = wave >> 4;          // / HH
    int h = wave & (HH - 1);
    if (l >= LL) return;
    int lane = threadIdx.x & 31;
    int d0 = lane * 2;
    const float* qrow = q + (size_t)l * DD + h * HD;
    float q0 = qrow[d0], q1 = qrow[d0 + 1];
    float mmax = -3.0e38f, ssum = 0.0f, a0 = 0.0f, a1 = 0.0f;
    int j0 = l - WIN + 1; if (j0 < 0) j0 = 0;
    for (int j = j0; j <= l; ++j) {
        const float* krow = k + (size_t)j * DD + h * HD;
        float p = q0 * krow[d0] + q1 * krow[d0 + 1];
#pragma unroll
        for (int off = 16; off >= 1; off >>= 1) p += __shfl_xor(p, off, 32);
        p *= 0.125f;  // HD^-0.5
        float mn = fmaxf(mmax, p);
        float corr = __expf(mmax - mn);
        float e = __expf(p - mn);
        const float* vrow = v + (size_t)j * DD + h * HD;
        a0 = a0 * corr + e * vrow[d0];
        a1 = a1 * corr + e * vrow[d0 + 1];
        ssum = ssum * corr + e;
        mmax = mn;
    }
    float o0 = a0 / ssum, o1 = a1 / ssum;
    if (!__builtin_isfinite(o0)) o0 = 0.0f;
    if (!__builtin_isfinite(o1)) o1 = 0.0f;
    bf16* orow = out_bf + (size_t)l * DD + h * HD;
    orow[d0] = (bf16)o0;
    orow[d0 + 1] = (bf16)o1;
}

// =====================================================================
// SwiGLU: h = silu(z1) * z3, written bf16 into K-padded (L x IPAD) buf
// =====================================================================
__global__ __launch_bounds__(256)
void swiglu_k(const float* __restrict__ z1, const float* __restrict__ z3,
              bf16* __restrict__ hb, long long total) {
    long long i = (long long)blockIdx.x * 256 + threadIdx.x;
    if (i >= total) return;
    long long r = i / IPAD;
    int c = (int)(i - r * IPAD);
    float o = 0.0f;
    if (c < II) {
        float a = z1[r * (long long)II + c];
        float b = z3[r * (long long)II + c];
        o = (a / (1.0f + __expf(-a))) * b;
    }
    hb[i] = (bf16)o;
}

// =====================================================================
// WMMA bf16 NT-GEMM: C[M,N] = A[M,K] * Bw[N,K]^T + bias (+resid)(clamp)
// Register-blocked: each wave owns a 32x64 C macro-tile (2x4 16x16 tiles,
// 8 f32x8 accumulators). Per 32-wide K step: 4 A-fragment b128 loads +
// 8 B-fragment b128 loads feed 8 v_wmma_f32_16x16x32_bf16 (1.5 loads/wmma,
// vs 4 for the unblocked version) -> 4x less A traffic, 2x less B traffic
// through L2 (all bf16 weights resident in the 192MB L2).
// N-tile count may not be a multiple of NB: loads clamp to the last tile
// (wave-uniform, EXEC stays all-ones through every WMMA), stores guard.
// mode bit0: add residual (read C-shaped resid), bit1: clamp to +-50.
// =====================================================================
__global__ __launch_bounds__(256)
void gemm_bf16_wmma(const bf16* __restrict__ A, const bf16* __restrict__ Bw,
                    const float* __restrict__ bias, const float* __restrict__ resid,
                    float* __restrict__ C, int M, int N, int K, int mode) {
    int ntiles = N >> 4;
    int ngrp = (ntiles + NB - 1) / NB;
    int mgrp = M >> 5;                     // M is a multiple of 32
    int wave = blockIdx.x * 8 + (threadIdx.x >> 5);
    if (wave >= mgrp * ngrp) return;       // wave-uniform exit
    int mg = wave / ngrp;
    int ng = wave - mg * ngrp;
    int lane = threadIdx.x & 31;
    int half = lane >> 4;
    int r = lane & 15;

    const bf16* Arow[MB];
#pragma unroll
    for (int i = 0; i < MB; ++i)
        Arow[i] = A + (size_t)(mg * 32 + i * 16 + r) * K;
    const bf16* Brow[NB];
#pragma unroll
    for (int j = 0; j < NB; ++j) {
        int nt = ng * NB + j;
        if (nt > ntiles - 1) nt = ntiles - 1;   // clamp (uniform per wave)
        Brow[j] = Bw + (size_t)(nt * 16 + r) * K;
    }

    f32x8 acc[MB][NB];
#pragma unroll
    for (int i = 0; i < MB; ++i)
#pragma unroll
        for (int j = 0; j < NB; ++j)
            acc[i][j] = (f32x8){0.f, 0.f, 0.f, 0.f, 0.f, 0.f, 0.f, 0.f};

    for (int k0 = 0; k0 < K; k0 += 32) {
        bf16x16 afrag[MB];
#pragma unroll
        for (int i = 0; i < MB; ++i) {
            // lane holds K in [k0+8*half, +8) and [k0+16+8*half, +8)
            bf16x8 alo = *(const bf16x8*)(Arow[i] + k0 + 8 * half);
            bf16x8 ahi = *(const bf16x8*)(Arow[i] + k0 + 16 + 8 * half);
            afrag[i] = __builtin_shufflevector(alo, ahi,
                           0, 1, 2, 3, 4, 5, 6, 7, 8, 9, 10, 11, 12, 13, 14, 15);
        }
        bf16x16 bfrag[NB];
#pragma unroll
        for (int j = 0; j < NB; ++j) {
            // lane (= column) holds K in [k0+8*half, +8) and [k0+16+8*half, +8)
            bf16x8 blo = *(const bf16x8*)(Brow[j] + k0 + 8 * half);
            bf16x8 bhi = *(const bf16x8*)(Brow[j] + k0 + 16 + 8 * half);
            bfrag[j] = __builtin_shufflevector(blo, bhi,
                           0, 1, 2, 3, 4, 5, 6, 7, 8, 9, 10, 11, 12, 13, 14, 15);
        }
#pragma unroll
        for (int i = 0; i < MB; ++i)
#pragma unroll
            for (int j = 0; j < NB; ++j)
                acc[i][j] = __builtin_amdgcn_wmma_f32_16x16x32_bf16(
                                false, afrag[i], false, bfrag[j],
                                (short)0, acc[i][j], false, false);
    }

    // epilogue: C layout per tile: VGPR v -> row v + 8*half, col = lane&15
#pragma unroll
    for (int j = 0; j < NB; ++j) {
        int nt = ng * NB + j;
        if (nt >= ntiles) continue;        // uniform; only after all WMMA
        int n = nt * 16 + r;
        float bv = bias[n];
#pragma unroll
        for (int i = 0; i < MB; ++i) {
#pragma unroll
            for (int v = 0; v < 8; ++v) {
                int m = mg * 32 + i * 16 + 8 * half + v;
                size_t idx = (size_t)m * N + n;
                float o = acc[i][j][v] + bv;
                if (mode & 1) o += resid[idx];
                if (mode & 2) o = fminf(fmaxf(o, -50.0f), 50.0f);
                C[idx] = o;
            }
        }
    }
}

// =====================================================================
// Host orchestration
// =====================================================================
static inline int blk1d(long long total) { return (int)((total + 255) / 256); }

extern "C" void kernel_launch(void* const* d_in, const int* in_sizes, int n_in,
                              void* d_out, int out_size, void* d_ws, size_t ws_size,
                              hipStream_t stream) {
    // ---- inputs (setup_inputs order) ----
    const int*   tokens  = (const int*)  d_in[0];
    const float* embed_w = (const float*)d_in[1];
    const float* ln1_w   = (const float*)d_in[2];
    const float* ln2_w   = (const float*)d_in[3];
    const float* wq_w    = (const float*)d_in[4];
    const float* wq_b    = (const float*)d_in[5];
    const float* wk_w    = (const float*)d_in[6];
    const float* wk_b    = (const float*)d_in[7];
    const float* wv_w    = (const float*)d_in[8];
    const float* wv_b    = (const float*)d_in[9];
    const float* wo_w    = (const float*)d_in[10];
    const float* wo_b    = (const float*)d_in[11];
    const float* qn_w    = (const float*)d_in[12];
    const float* kn_w    = (const float*)d_in[13];
    const float* w1_w    = (const float*)d_in[14];
    const float* w1_b    = (const float*)d_in[15];
    const float* w2_w    = (const float*)d_in[16];
    const float* w2_b    = (const float*)d_in[17];
    const float* w3_w    = (const float*)d_in[18];
    const float* w3_b    = (const float*)d_in[19];
    const float* lnf_w   = (const float*)d_in[20];
    const float* lm_w    = (const float*)d_in[21];
    const float* lm_b    = (const float*)d_in[22];
    float* logits = (float*)d_out;

    // ---- workspace carve-up (256B aligned) ----
    size_t off = 0;
    auto alloc = [&](size_t bytes) -> void* {
        void* p = (char*)d_ws + off;
        off = (off + bytes + 255) & ~(size_t)255;
        return p;
    };
    float* x       = (float*)alloc((size_t)LL * DD * 4);       // residual stream
    bf16*  xn_bf   = (bf16*) alloc((size_t)LL * DD * 2);       // normed activ (GEMM A)
    float* qb      = (float*)alloc((size_t)LL * DD * 4);
    float* kb      = (float*)alloc((size_t)LL * DD * 4);
    float* vb      = (float*)alloc((size_t)LL * DD * 4);
    bf16*  attn_bf = (bf16*) alloc((size_t)LL * DD * 2);
    float* z1      = (float*)alloc((size_t)LL * II * 4);
    float* z3      = (float*)alloc((size_t)LL * II * 4);
    bf16*  h_bf    = (bf16*) alloc((size_t)LL * IPAD * 2);
    // per-layer bf16 weights (reused each layer)
    bf16* wq_bf = (bf16*)alloc((size_t)DD * DD * 2);
    bf16* wk_bf = (bf16*)alloc((size_t)DD * DD * 2);
    bf16* wv_bf = (bf16*)alloc((size_t)DD * DD * 2);
    bf16* wo_bf = (bf16*)alloc((size_t)DD * DD * 2);
    bf16* w1_bf = (bf16*)alloc((size_t)II * DD * 2);
    bf16* w3_bf = (bf16*)alloc((size_t)II * DD * 2);
    bf16* w2_bf = (bf16*)alloc((size_t)DD * IPAD * 2);
    bf16* lm_bf = (bf16*)alloc((size_t)VV * DD * 2);

    auto cvt = [&](const float* src, bf16* dst, long long rows, int K, int Kp) {
        long long total = rows * Kp;
        cvt_bf16_pad<<<blk1d(total), 256, 0, stream>>>(src, dst, K, Kp, total);
    };
    auto gemm = [&](const bf16* A, const bf16* Bw, const float* bias,
                    const float* resid, float* C, int M, int N, int K, int mode) {
        int ntiles = N >> 4;
        int waves = (M >> 5) * ((ntiles + NB - 1) / NB);
        gemm_bf16_wmma<<<(waves + 7) / 8, 256, 0, stream>>>(A, Bw, bias, resid, C,
                                                            M, N, K, mode);
    };

    // ---- embed ----
    embed_k<<<blk1d((long long)LL * DD), 256, 0, stream>>>(
        tokens, embed_w, x, (long long)LL * DD);

    // ---- layers ----
    for (int l = 0; l < NLYR; ++l) {
        const size_t dd = (size_t)l * DD * DD;
        const size_t id = (size_t)l * II * DD;
        // convert this layer's weights to bf16
        cvt(wq_w + dd, wq_bf, DD, DD, DD);
        cvt(wk_w + dd, wk_bf, DD, DD, DD);
        cvt(wv_w + dd, wv_bf, DD, DD, DD);
        cvt(wo_w + dd, wo_bf, DD, DD, DD);
        cvt(w1_w + id, w1_bf, II, DD, DD);
        cvt(w3_w + id, w3_bf, II, DD, DD);
        cvt(w2_w + (size_t)l * DD * II, w2_bf, DD, II, IPAD);

        // attention block
        rmsnorm_row<<<LL, 256, 0, stream>>>(x, ln1_w + (size_t)l * DD, xn_bf);
        gemm(xn_bf, wq_bf, wq_b + (size_t)l * DD, nullptr, qb, LL, DD, DD, 0);
        gemm(xn_bf, wk_bf, wk_b + (size_t)l * DD, nullptr, kb, LL, DD, DD, 0);
        gemm(xn_bf, wv_bf, wv_b + (size_t)l * DD, nullptr, vb, LL, DD, DD, 0);
        int hrows = LL * HH;
        head_rmsnorm<<<(hrows + 7) / 8, 256, 0, stream>>>(qb, qn_w + (size_t)l * HD, hrows);
        head_rmsnorm<<<(hrows + 7) / 8, 256, 0, stream>>>(kb, kn_w + (size_t)l * HD, hrows);
        attn_sw<<<(LL * HH + 7) / 8, 256, 0, stream>>>(qb, kb, vb, attn_bf);
        gemm(attn_bf, wo_bf, wo_b + (size_t)l * DD, x, x, LL, DD, DD, 1); // +residual

        // MLP block
        rmsnorm_row<<<LL, 256, 0, stream>>>(x, ln2_w + (size_t)l * DD, xn_bf);
        gemm(xn_bf, w1_bf, w1_b + (size_t)l * II, nullptr, z1, LL, II, DD, 0);
        gemm(xn_bf, w3_bf, w3_b + (size_t)l * II, nullptr, z3, LL, II, DD, 0);
        swiglu_k<<<blk1d((long long)LL * IPAD), 256, 0, stream>>>(
            z1, z3, h_bf, (long long)LL * IPAD);
        gemm(h_bf, w2_bf, w2_b + (size_t)l * DD, x, x, LL, DD, IPAD, 1); // +residual
    }

    // ---- final norm + LM head (bias + clamp +-50 fused) ----
    rmsnorm_row<<<LL, 256, 0, stream>>>(x, lnf_w, xn_bf);
    cvt(lm_w, lm_bf, VV, DD, DD);
    gemm(xn_bf, lm_bf, lm_b, nullptr, logits, LL, VV, DD, 2);
}
